// MOCO_58540404244866
// MI455X (gfx1250) — matrix operand order repaired and checked by reading, not compiled, not run on hardware
//
#include <hip/hip_runtime.h>
#include <hip/hip_bf16.h>

typedef __attribute__((ext_vector_type(16))) _Float16 v16h;
typedef __attribute__((ext_vector_type(8)))  _Float16 v8h;
typedef __attribute__((ext_vector_type(8)))  float    v8f;

#define LEAKY(v) ((v) > 0.f ? (v) : 0.1f * (v))

// ---------------------------------------------------------------------------
// elementwise helpers
// ---------------------------------------------------------------------------
__global__ void momentum_axpy(const float* __restrict__ kp, const float* __restrict__ qp,
                              float* __restrict__ o, int n) {
    int i = blockIdx.x * blockDim.x + threadIdx.x;
    if (i < n) o[i] = 0.999f * kp[i] + 0.001f * qp[i];
}

__global__ void f32_to_f16(const float* __restrict__ x, _Float16* __restrict__ y, long n) {
    long i = blockIdx.x * (long)blockDim.x + threadIdx.x;
    long gs = (long)gridDim.x * blockDim.x;
    for (; i < n; i += gs) y[i] = (_Float16)x[i];
}

// transpose+convert weights: w[Ktot][Cout] f32 -> wt[Cout][Kpad] f16 (zero K-pad)
__global__ void w_to_f16T(const float* __restrict__ w, _Float16* __restrict__ wt,
                          int Ktot, int Cout, int Kpad) {
    long n = (long)Cout * Kpad;
    long i = blockIdx.x * (long)blockDim.x + threadIdx.x;
    long gs = (long)gridDim.x * blockDim.x;
    for (; i < n; i += gs) {
        int c = (int)(i / Kpad);
        int k = (int)(i - (long)c * Kpad);
        wt[i] = (k < Ktot) ? (_Float16)w[(long)k * Cout + c] : (_Float16)0.f;
    }
}

// ---------------------------------------------------------------------------
// implicit-GEMM conv via WMMA: Y[m][n] = sum_k A(im2col)[m][k] * W[k][n] + bias[n]
// block = 128 threads (4 waves); one 16-row M tile shared via LDS, 4x16 channels.
// Weights pre-transposed to [Cout][Kpad] so each lane loads B as 2x16B vectors.
// ---------------------------------------------------------------------------
__global__ void conv_wmma(const _Float16* __restrict__ X, const _Float16* __restrict__ Wt,
                          const float* __restrict__ bias, _Float16* __restrict__ Y,
                          int Bn, int H, int W, int Cin, int Ho, int Wo, int Cout,
                          int stride, int padBeg, int Ktot, int Kpad) {
    __shared__ __align__(16) _Float16 As[16][32];
    const int lane = threadIdx.x & 31;
    const int wave = threadIdx.x >> 5;
    const int m0   = blockIdx.x * 16;
    const int n0   = (blockIdx.y * 4 + wave) * 16;
    const int mrow = lane & 15;
    const int hf   = lane >> 4;
    const bool fast = (Cin & 31) == 0;   // 32-wide K window contiguous in cin

    const _Float16* brow0 = Wt + (long)(n0 + mrow) * Kpad + hf * 16;

    v8f acc = {};
    for (int kk = 0; kk < Kpad; kk += 32) {
        __syncthreads();
        if (fast) {
            if (threadIdx.x < 64) {       // 16 rows x 4 chunks of 8 halfs
                int r = threadIdx.x >> 2, ch = threadIdx.x & 3;
                int m   = m0 + r;
                int b   = m / (Ho * Wo);
                int rem = m - b * (Ho * Wo);
                int ho  = rem / Wo, wo = rem - ho * Wo;
                int khw  = kk / Cin;
                int cinB = kk - khw * Cin;
                int kh = khw / 3, kw = khw - kh * 3;
                int hi = ho * stride + kh - padBeg;
                int wi = wo * stride + kw - padBeg;
                v8h v = {};
                if (hi >= 0 && hi < H && wi >= 0 && wi < W)
                    v = *(const v8h*)&X[((long)(b * H + hi) * W + wi) * Cin + cinB + ch * 8];
                *(v8h*)&As[r][ch * 8] = v;
            }
        } else {                          // generic per-element im2col (layer 0)
            for (int idx = threadIdx.x; idx < 512; idx += 128) {
                int r = idx >> 5, kl = idx & 31;
                int k = kk + kl;
                _Float16 v = (_Float16)0.f;
                if (k < Ktot) {
                    int m   = m0 + r;
                    int b   = m / (Ho * Wo);
                    int rem = m - b * (Ho * Wo);
                    int ho  = rem / Wo, wo = rem - ho * Wo;
                    int cin = k % Cin;
                    int khw = k / Cin;
                    int kh  = khw / 3, kw = khw - kh * 3;
                    int hi  = ho * stride + kh - padBeg;
                    int wi  = wo * stride + kw - padBeg;
                    if (hi >= 0 && hi < H && wi >= 0 && wi < W)
                        v = X[((long)(b * H + hi) * W + wi) * Cin + cin];
                }
                As[r][kl] = v;
            }
        }
        __syncthreads();
        // A fragment: lane half hf holds k = 16*hf..16*hf+15 of row mrow (contiguous, same as B)
        v8h alo = *(const v8h*)&As[mrow][hf * 16];
        v8h ahi = *(const v8h*)&As[mrow][hf * 16 + 8];
        // B fragment: lane holds K=kb..kb+15 for column n (contiguous in Wt)
        const _Float16* brow = brow0 + kk;
        v8h b0 = *(const v8h*)brow;
        v8h b1 = *(const v8h*)(brow + 8);
        if (kk + 32 < Kpad) __builtin_prefetch(brow + 32, 0, 3);
        v16h a, bf;
#pragma unroll
        for (int i = 0; i < 8; ++i) {
            a[i] = alo[i];  a[i + 8] = ahi[i];
            bf[i] = b0[i];  bf[i + 8] = b1[i];
        }
        acc = __builtin_amdgcn_wmma_f32_16x16x32_f16(false, a, false, bf,
                                                     (short)0, acc, false, false);
    }
    int n = n0 + mrow;
    float bv = bias[n];
#pragma unroll
    for (int v = 0; v < 8; ++v) {
        int m = m0 + v + hf * 8;
        Y[(long)m * Cout + n] = (_Float16)(acc[v] + bv);
    }
}

// ---------------------------------------------------------------------------
// generic WMMA GEMM: C = epi(A @ Bt^T); A[M][K] f16, Bt[N][ldb] f16 (row = out col)
// ---------------------------------------------------------------------------
__global__ void fc_wmma(const _Float16* __restrict__ A, const _Float16* __restrict__ Bt,
                        const float* __restrict__ bias, float* __restrict__ C,
                        int M, int N, int K, long ldb, long ldc, long col0,
                        float scale, int leaky) {
    const int lane = threadIdx.x & 31;
    const int m0 = blockIdx.x * 16, n0 = blockIdx.y * 16;
    const int mrow = lane & 15, hf = lane >> 4;
    const _Float16* arow0 = A + (long)(m0 + mrow) * K;
    const _Float16* brow0 = Bt + (long)(n0 + mrow) * ldb + hf * 16;
    v8f acc = {};
    for (int kk = 0; kk < K; kk += 32) {
        v8h a0 = *(const v8h*)(arow0 + kk + hf * 16);
        v8h a1 = *(const v8h*)(arow0 + kk + hf * 16 + 8);
        v8h b0 = *(const v8h*)(brow0 + kk);
        v8h b1 = *(const v8h*)(brow0 + kk + 8);
        v16h a, bf;
#pragma unroll
        for (int i = 0; i < 8; ++i) {
            a[i] = a0[i];  a[i + 8] = a1[i];
            bf[i] = b0[i]; bf[i + 8] = b1[i];
        }
        acc = __builtin_amdgcn_wmma_f32_16x16x32_f16(false, a, false, bf,
                                                     (short)0, acc, false, false);
    }
    int n = n0 + mrow;
    float bv = bias ? bias[n] : 0.f;
#pragma unroll
    for (int v = 0; v < 8; ++v) {
        int m = m0 + v + hf * 8;
        float val = acc[v] + bv;
        if (leaky) val = LEAKY(val);
        C[(long)m * ldc + col0 + n] = val * scale;
    }
}

// ---------------------------------------------------------------------------
// batch-norm: per-channel stats then fused scale/shift + leaky (f16 out)
// ---------------------------------------------------------------------------
__global__ void bn_stats(const _Float16* __restrict__ Y, const float* __restrict__ g,
                         const float* __restrict__ b, float* __restrict__ scl,
                         float* __restrict__ sft, int Mrows, int Cout) {
    int c = blockIdx.x;
    float s = 0.f, s2 = 0.f;
    for (int r = threadIdx.x; r < Mrows; r += 256) {
        float v = (float)Y[(long)r * Cout + c];
        s += v; s2 += v * v;
    }
    __shared__ float sh[256], sh2[256];
    sh[threadIdx.x] = s; sh2[threadIdx.x] = s2;
    __syncthreads();
    for (int off = 128; off > 0; off >>= 1) {
        if (threadIdx.x < off) {
            sh[threadIdx.x]  += sh[threadIdx.x + off];
            sh2[threadIdx.x] += sh2[threadIdx.x + off];
        }
        __syncthreads();
    }
    if (threadIdx.x == 0) {
        float mu  = sh[0] / Mrows;
        float var = sh2[0] / Mrows - mu * mu;
        float sc  = g[c] * rsqrtf(var + 1e-3f);
        scl[c] = sc;
        sft[c] = b[c] - mu * sc;
    }
}

__global__ void bn_apply(const _Float16* __restrict__ Y, const float* __restrict__ scl,
                         const float* __restrict__ sft, _Float16* __restrict__ O,
                         long n, int Cout) {
    long i = blockIdx.x * (long)blockDim.x + threadIdx.x;
    long gs = (long)gridDim.x * blockDim.x;
    for (; i < n; i += gs) {
        int c = (int)(i % Cout);
        float v = (float)Y[i] * scl[c] + sft[c];
        O[i] = (_Float16)LEAKY(v);
    }
}

// ---------------------------------------------------------------------------
// global average pool, l2norm, l_pos/labels, queue shift
// ---------------------------------------------------------------------------
__global__ void pool_mean(const _Float16* __restrict__ X, float* __restrict__ F,
                          float* __restrict__ F2, int Bn, int HW, int C) {
    int idx = blockIdx.x * blockDim.x + threadIdx.x;
    if (idx >= Bn * C) return;
    int b = idx / C, c = idx - b * C;
    const _Float16* p = X + (long)b * HW * C + c;
    float s = 0.f;
    for (int i = 0; i < HW; ++i) s += (float)p[(long)i * C];
    float m = s / HW;
    F[idx] = m;
    if (F2) F2[idx] = m;
}

__global__ void l2norm_rows(const float* __restrict__ X, float* __restrict__ Y,
                            _Float16* __restrict__ Yh, int ncols) {
    int row = blockIdx.x;
    const float* x = X + (long)row * ncols;
    float s = 0.f;
    for (int i = threadIdx.x; i < ncols; i += 256) s += x[i] * x[i];
    __shared__ float sh[256];
    sh[threadIdx.x] = s;
    __syncthreads();
    for (int off = 128; off > 0; off >>= 1) {
        if (threadIdx.x < off) sh[threadIdx.x] += sh[threadIdx.x + off];
        __syncthreads();
    }
    float inv = rsqrtf(fmaxf(sh[0], 1e-12f));
    for (int i = threadIdx.x; i < ncols; i += 256) {
        float v = x[i] * inv;
        Y[(long)row * ncols + i] = v;
        if (Yh) Yh[(long)row * ncols + i] = (_Float16)v;
    }
}

__global__ void lpos_labels(const float* __restrict__ qn, const float* __restrict__ kn,
                            float* __restrict__ lg, float* __restrict__ lb,
                            int Bn, int feat, long ldl, float invT) {
    int b = threadIdx.x;
    if (b < Bn) {
        float s = 0.f;
        for (int i = 0; i < feat; ++i) s += qn[b * feat + i] * kn[b * feat + i];
        lg[(long)b * ldl] = s * invT;
        lb[b] = 0.f;
    }
}

__global__ void queue_update(const float* __restrict__ q, const float* __restrict__ kn,
                             float* __restrict__ o, long total, long keep) {
    long i = blockIdx.x * (long)blockDim.x + threadIdx.x;
    long gs = (long)gridDim.x * blockDim.x;
    for (; i < total; i += gs)
        o[i] = (i < keep) ? q[i + (total - keep)] : kn[i - keep];
}

// ---------------------------------------------------------------------------
// host orchestration
// ---------------------------------------------------------------------------
static inline int gblocks(long n, int bs, long cap) {
    long g = (n + bs - 1) / bs;
    return (int)(g < cap ? g : cap);
}

extern "C" void kernel_launch(void* const* d_in, const int* in_sizes, int n_in,
                              void* d_out, int out_size, void* d_ws, size_t ws_size,
                              hipStream_t stream) {
    (void)in_sizes; (void)n_in; (void)out_size; (void)ws_size;
    const int BATCH = 32, FEAT = 256, KQ = 8192;
    static const int  Cin_[6]  = {3, 64, 64, 128, 128, 256};
    static const int  Cout_[6] = {64, 64, 128, 128, 256, 256};
    static const int  Hin_[6]  = {128, 128, 128, 64, 64, 32};
    static const int  Str_[6]  = {1, 1, 2, 1, 2, 1};
    static const int  Hout_[6] = {128, 128, 64, 64, 32, 32};
    // alphabetical leaf order of the params pytree
    static const int leafSize[28] = {
        64, 64, 64, 64, 128, 128, 128, 128, 256, 256, 256, 256,        // bn{0..5}_{b,g}
        64, 9*3*64, 64, 9*64*64, 128, 9*64*128, 128, 9*128*128,        // conv0..3 b,w
        256, 9*128*256, 256, 9*256*256,                                 // conv4,5 b,w
        256, 65536, 256, 65536};                                        // dense0,1 b,w

    const float* inputs = (const float*)d_in[0];
    const float* qleaf[28];
    const float* kleaf_in[28];
    for (int j = 0; j < 28; ++j) { qleaf[j] = (const float*)d_in[1 + j];
                                   kleaf_in[j] = (const float*)d_in[29 + j]; }
    const float* queue = (const float*)d_in[57];
    float* out = (float*)d_out;

    const long OFF_FEAT = 0, OFF_LOG = (long)BATCH * FEAT;
    const long OFF_LAB = OFF_LOG + (long)BATCH * (KQ + 1);
    const long OFF_NQ  = OFF_LAB + BATCH;

    // ---- workspace bump allocator ----
    char* base = (char*)d_ws;
    size_t cur = 0;
    auto alloc = [&](size_t bytes) -> void* {
        void* p = base + cur;
        cur = (cur + bytes + 255) & ~(size_t)255;
        return p;
    };
    _Float16* bufA   = (_Float16*)alloc(33554432ull * 2);
    _Float16* bufB   = (_Float16*)alloc(33554432ull * 2);
    _Float16* wtF16  = (_Float16*)alloc(589824ull * 2);           // [Cout][Kpad]
    _Float16* queueH = (_Float16*)alloc((size_t)KQ * FEAT * 2);   // already [N][K]
    float* scl   = (float*)alloc(256 * 4);
    float* sft   = (float*)alloc(256 * 4);
    float* feats = (float*)alloc(BATCH * FEAT * 4);
    _Float16* featsH = (_Float16*)alloc(BATCH * FEAT * 2);
    float* hbuf  = (float*)alloc(BATCH * FEAT * 4);
    _Float16* hH = (_Float16*)alloc(BATCH * FEAT * 2);
    float* qout  = (float*)alloc(BATCH * FEAT * 4);
    float* kout  = (float*)alloc(BATCH * FEAT * 4);
    float* qn    = (float*)alloc(BATCH * FEAT * 4);
    float* kn    = (float*)alloc(BATCH * FEAT * 4);
    _Float16* qnH = (_Float16*)alloc(BATCH * FEAT * 2);
    float* kleaf[28];
    for (int j = 0; j < 28; ++j) kleaf[j] = (float*)alloc((size_t)leafSize[j] * 4);

    // ---- 1) momentum update of key encoder params into workspace ----
    for (int j = 0; j < 28; ++j)
        momentum_axpy<<<gblocks(leafSize[j], 256, 1 << 20), 256, 0, stream>>>(
            kleaf_in[j], qleaf[j], kleaf[j], leafSize[j]);

    // ---- encoder runner ----
    auto run_encoder = [&](const float* const* L, float* featDst2, float* outPtr) {
        long nin = (long)BATCH * 128 * 128 * 3;
        f32_to_f16<<<gblocks(nin, 256, 8192), 256, 0, stream>>>(inputs, bufA, nin);
        for (int i = 0; i < 6; ++i) {
            int Cin = Cin_[i], Cout = Cout_[i], H = Hin_[i], Ho = Hout_[i], st = Str_[i];
            int Ktot = 9 * Cin, pad = (st == 1) ? 1 : 0;
            int Kpad = (Ktot + 31) & ~31;
            long M = (long)BATCH * Ho * Ho;
            const float* cw = L[13 + 2 * i];
            const float* cb = L[12 + 2 * i];
            const float* bg = L[2 * i + 1];
            const float* bb = L[2 * i];
            long nw = (long)Cout * Kpad;
            w_to_f16T<<<gblocks(nw, 256, 4096), 256, 0, stream>>>(cw, wtF16, Ktot, Cout, Kpad);
            dim3 grid((unsigned)(M / 16), (unsigned)(Cout / 64));
            conv_wmma<<<grid, 128, 0, stream>>>(bufA, wtF16, cb, bufB,
                                                BATCH, H, H, Cin, Ho, Ho, Cout, st, pad,
                                                Ktot, Kpad);
            bn_stats<<<Cout, 256, 0, stream>>>(bufB, bg, bb, scl, sft, (int)M, Cout);
            long nact = M * Cout;
            bn_apply<<<gblocks(nact, 256, 8192), 256, 0, stream>>>(bufB, scl, sft, bufA, nact, Cout);
        }
        // global average pool: bufA is [32,32,32,256]
        pool_mean<<<gblocks(BATCH * FEAT, 256, 1024), 256, 0, stream>>>(
            bufA, feats, featDst2, BATCH, 32 * 32, FEAT);
        f32_to_f16<<<gblocks(BATCH * FEAT, 256, 64), 256, 0, stream>>>(feats, featsH, BATCH * FEAT);
        // dense0 (+leaky): transpose W -> [N][K]
        w_to_f16T<<<gblocks(FEAT * FEAT, 256, 512), 256, 0, stream>>>(L[25], wtF16, FEAT, FEAT, FEAT);
        fc_wmma<<<dim3(BATCH / 16, FEAT / 16), 32, 0, stream>>>(
            featsH, wtF16, L[24], hbuf, BATCH, FEAT, FEAT, FEAT, FEAT, 0, 1.0f, 1);
        f32_to_f16<<<gblocks(BATCH * FEAT, 256, 64), 256, 0, stream>>>(hbuf, hH, BATCH * FEAT);
        // dense1
        w_to_f16T<<<gblocks(FEAT * FEAT, 256, 512), 256, 0, stream>>>(L[27], wtF16, FEAT, FEAT, FEAT);
        fc_wmma<<<dim3(BATCH / 16, FEAT / 16), 32, 0, stream>>>(
            hH, wtF16, L[26], outPtr, BATCH, FEAT, FEAT, FEAT, FEAT, 0, 1.0f, 0);
    };

    // ---- 2) query encoder (features -> d_out), 3) key encoder ----
    run_encoder(qleaf, out + OFF_FEAT, qout);
    const float* kl[28];
    for (int j = 0; j < 28; ++j) kl[j] = kleaf[j];
    run_encoder(kl, nullptr, kout);

    // ---- 4) l2 normalize q (also f16 copy) and k ----
    l2norm_rows<<<BATCH, 256, 0, stream>>>(qout, qn, qnH, FEAT);
    l2norm_rows<<<BATCH, 256, 0, stream>>>(kout, kn, nullptr, FEAT);

    // ---- 5) l_pos (logits col 0) + labels ----
    lpos_labels<<<1, 32, 0, stream>>>(qn, kn, out + OFF_LOG, out + OFF_LAB,
                                      BATCH, FEAT, KQ + 1, 1.0f / 0.07f);

    // ---- 6) l_neg = qn @ queue^T / T : queue rows are already the GEMM "Bt" layout
    f32_to_f16<<<gblocks((long)KQ * FEAT, 256, 8192), 256, 0, stream>>>(
        queue, queueH, (long)KQ * FEAT);
    fc_wmma<<<dim3(BATCH / 16, KQ / 16), 32, 0, stream>>>(
        qnH, queueH, nullptr, out + OFF_LOG, BATCH, KQ, FEAT, FEAT, KQ + 1, 1, 1.0f / 0.07f, 0);

    // ---- 7) circular queue update ----
    long total = (long)KQ * FEAT, keep = (long)(KQ - BATCH) * FEAT;
    queue_update<<<gblocks(total, 256, 8192), 256, 0, stream>>>(
        queue, kn, out + OFF_NQ, total, keep);
}
